// SNN_Model_22814866276709
// MI455X (gfx1250) — matrix operand order repaired and checked by reading, not compiled, not run on hardware
//
#include <hip/hip_runtime.h>
#include <hip/hip_bf16.h>
#include <cmath>

// ---------------------------------------------------------------------------
// SNN (3-step) fused bf16-WMMA implementation for gfx1250 (MI455X).
// - all big GEMMs on v_wmma_f32_16x16x32_bf16 (fp32 accum, fused epilogues)
// - B operands stored transposed (N-major): A and B tiles are direct 2D tiles
//   staged by TDM tensor_load_to_lds (pad feature produces padded LDS rows)
// - BK=64 K-steps: 16 WMMAs per barrier/tensor-wait, 16KB TDM tiles
// - double-buffered LDS: TDM for tile t+1 overlaps wmma compute of tile t
// ---------------------------------------------------------------------------

typedef __attribute__((ext_vector_type(16))) __bf16 v16bf;
typedef __attribute__((ext_vector_type(8)))  __bf16 v8bf;
typedef __attribute__((ext_vector_type(8)))  unsigned short u16x8;
typedef __attribute__((ext_vector_type(8)))  float  v8f;

#define BM 128
#define BN 128
#define BK 64
#define LDSW 72   // BK + 8 pad (bf16 elems); 144B row stride, 16B aligned

#define HAVE_TDM __has_builtin(__builtin_amdgcn_tensor_load_to_lds)

#if HAVE_TDM
typedef __attribute__((ext_vector_type(4))) unsigned int u32x4;
typedef __attribute__((ext_vector_type(8))) int i32x8;
typedef __attribute__((ext_vector_type(4))) int i32x4;

// Issue a 2D-tile TDM load: tile_d0 bf16 elems contiguous per row, tile_d1
// rows with global stride stride_d0 (elems); LDS dest gets pad_amount DWORDs
// inserted every interval (codes per CDNA5 D# group1) -> padded LDS rows.
__device__ __forceinline__ void tdm_load_tile_2d(
    const void* gaddr, unsigned lds_byte_addr,
    unsigned tensor_d0, unsigned tensor_d1, unsigned stride_d0,
    unsigned tile_d0, unsigned tile_d1,
    unsigned pad_interval_code, unsigned pad_amount_code) {
    unsigned long long ga = (unsigned long long)gaddr;
    u32x4 g0;
    g0[0] = 1u;                                             // count=1 (valid user D#)
    g0[1] = lds_byte_addr;                                  // lds_addr
    g0[2] = (unsigned)(ga & 0xffffffffu);                   // global_addr lo
    g0[3] = (unsigned)((ga >> 32) & 0x01ffffffu)            // global_addr hi (57-bit)
          | (2u << 30);                                     // type = 2 ("image")
    i32x8 g1;
    unsigned w0 = (1u << 16)                                // data_size = 2 bytes
                | (1u << 20)                                // pad_enable
                | (pad_interval_code << 22)
                | (pad_amount_code << 25);
    g1[0] = (int)w0;
    g1[1] = (int)((tensor_d0 & 0xffffu) << 16);             // tensor_dim0[15:0]
    g1[2] = (int)((tensor_d0 >> 16) & 0xffffu)              // tensor_dim0[31:16]
          | (int)((tensor_d1 & 0xffffu) << 16);             // tensor_dim1[15:0]
    g1[3] = (int)((tensor_d1 >> 16) & 0xffffu)              // tensor_dim1[31:16]
          | (int)((tile_d0 & 0xffffu) << 16);               // tile_dim0
    g1[4] = (int)(tile_d1 & 0xffffu);                       // tile_dim1 (tile_dim2=0)
    g1[5] = (int)stride_d0;                                 // tensor_dim0_stride lo32
    g1[6] = 0;
    g1[7] = 0;
    i32x4 z4 = {0, 0, 0, 0};
#if defined(__clang_major__) && (__clang_major__ >= 23)
    i32x8 z8 = {0, 0, 0, 0, 0, 0, 0, 0};
    __builtin_amdgcn_tensor_load_to_lds(g0, g1, z4, z4, z8, 0);
#else
    __builtin_amdgcn_tensor_load_to_lds(g0, g1, z4, z4, 0);
#endif
}
#endif  // HAVE_TDM

// Direct (no-transpose) 128x64 tile stage: src row-major, 64 contiguous bf16
// per row. TDM when available (wave 0 issues), else vectorized manual copy.
__device__ __forceinline__ void stage_tile_direct(const __bf16* __restrict__ src, int ldsrc,
                                                  int rows_total, __bf16 (*dst)[LDSW], int tid) {
#if HAVE_TDM
    if ((tid >> 5) == 0) {
        // 64 bf16 = 32 DW per row -> pad_interval code 4; +4 DW pad (code 3)
        // -> 36 DW = 72-elem LDS rows == LDSW.
        tdm_load_tile_2d(src, (unsigned)(unsigned long long)(const void*)&dst[0][0],
                         (unsigned)ldsrc, (unsigned)rows_total, (unsigned)ldsrc,
                         BK, BM, /*interval 32DW*/ 4, /*pad 4DW*/ 3);
    }
#else
#pragma unroll
    for (int it = 0; it < 4; ++it) {
        int idx = it * 256 + tid;          // 0..1023
        int row = idx >> 3;                // 128 rows
        int ch  = idx & 7;                 // 8 chunks of 8 elems
        *(v8bf*)&dst[row][ch * 8] = *(const v8bf*)(src + (size_t)row * ldsrc + ch * 8);
    }
#endif
}

// Transpose stage for the hebb A-tiles: dst[m][k] = src[k*ldsrc + m], two
// K-adjacent bf16 packed per ds_store_b32; rows beyond Mreal zero-filled.
__device__ __forceinline__ void stage_tile_transpose(const __bf16* __restrict__ src,
                                                     int ldsrc, int gm_base, int Mreal,
                                                     __bf16 (*dst)[LDSW], int tid) {
#pragma unroll
    for (int it = 0; it < 2; ++it) {
        int idx = it * 256 + tid;          // 0..511
        int kp  = idx >> 4;                // 0..31 -> k = 2*kp
        int k   = kp * 2;
        int mc  = idx & 15;                // 16 chunks of 8 m
        int gm  = gm_base + mc * 8;
        const __bf16* s0 = src + (size_t)k * ldsrc + mc * 8;
        const __bf16* s1 = s0 + ldsrc;
        if (gm + 8 <= Mreal) {
            u16x8 v0 = *(const u16x8*)s0;
            u16x8 v1 = *(const u16x8*)s1;
#pragma unroll
            for (int e = 0; e < 8; ++e) {
                unsigned p = (unsigned)v0[e] | ((unsigned)v1[e] << 16);
                *(unsigned*)&dst[mc * 8 + e][k] = p;
            }
        } else {
#pragma unroll
            for (int e = 0; e < 8; ++e) {
                dst[mc * 8 + e][k]     = (gm + e < Mreal) ? s0[e] : (__bf16)0.0f;
                dst[mc * 8 + e][k + 1] = (gm + e < Mreal) ? s1[e] : (__bf16)0.0f;
            }
        }
    }
}

// ------------------------------- utility kernels ---------------------------

__global__ __launch_bounds__(256)
void fill0_kernel(uint4* __restrict__ p, long n16) {
    long i = (long)blockIdx.x * 256 + threadIdx.x;
    if (i < n16) {
        uint4 z; z.x = 0u; z.y = 0u; z.z = 0u; z.w = 0u;
        p[i] = z;
    }
}

// x (B x IN) f32 -> xb (B x INP) bf16, zero-padded columns IN..INP-1
__global__ __launch_bounds__(256)
void cvt_x_kernel(const float* __restrict__ x, __bf16* __restrict__ xb,
                  int IN, int INP, int total) {
    int idx = blockIdx.x * 256 + threadIdx.x;
    if (idx >= total) return;
    int row = idx / INP;
    int col = idx - row * INP;
    float v = (col < IN) ? x[(size_t)row * IN + col] : 0.0f;
    xb[idx] = (__bf16)v;
}

// fwT[j][i] = W[j][i]*mask[i][j] + alpha*hebb[i][j]*(1-mask[i][j]); cols >= Kin are 0.
// Stored N-major (H x Kpad) so GEMM B-tiles are direct 2D tiles for the TDM.
__global__ __launch_bounds__(256)
void prep_fastwT_kernel(const float* __restrict__ W, const float* __restrict__ mask,
                        const float* __restrict__ hebb, const float* __restrict__ alpha,
                        __bf16* __restrict__ fwT, int Kin, int Kpad, int H) {
    int idx = blockIdx.x * 256 + threadIdx.x;
    if (idx >= H * Kpad) return;
    int j = idx / Kpad;        // output unit
    int i = idx - j * Kpad;    // input unit
    float v = 0.0f;
    if (i < Kin) {
        float m = mask[(size_t)i * H + j];
        v = W[(size_t)j * Kin + i] * m + alpha[0] * hebb[(size_t)i * H + j] * (1.0f - m);
    }
    fwT[idx] = (__bf16)v;
}

// ------------------------------- WMMA GEMM ---------------------------------
// C(MxN) = A' @ B, with B supplied TRANSPOSED: BT is N-major (N x K, ldbt).
// TRANSA=false: A'[m,k] = A[m*lda+k];  TRANSA=true: A'[m,k] = A[k*lda+m].
// MODE 0 (state): SNN epilogue; scl=decay; vec=eta[N]; postT written N-major.
// MODE 1 (hebb):  S = 0.8*S + vec[m]*scl*acc; scl=decay/B; vec=beta[M].
template <bool TRANSA, int MODE>
__global__ __launch_bounds__(256)
void gemm_wmma_kernel(const __bf16* __restrict__ A, int lda,
                      const __bf16* __restrict__ BT, int ldbt, int K,
                      float* __restrict__ S,
                      __bf16* __restrict__ spike, __bf16* __restrict__ postT,
                      const float* __restrict__ vec, float scl,
                      int Mreal, int N) {
    __shared__ __align__(16) __bf16 As[2][BM][LDSW];
    __shared__ __align__(16) __bf16 Bs[2][BN][LDSW];

    const int tid  = threadIdx.x;
    const int lane = tid & 31;
    const int wave = tid >> 5;
    const int wr   = wave >> 1;   // 0..3
    const int wc   = wave & 1;    // 0..1
    const int sel  = lane >> 4;   // 0/1
    const int l16  = lane & 15;

    const int m0 = blockIdx.y * BM;
    const int n0 = blockIdx.x * BN;

    v8f c[2][4];
#pragma unroll
    for (int i = 0; i < 2; ++i)
#pragma unroll
        for (int j = 0; j < 4; ++j)
#pragma unroll
            for (int e = 0; e < 8; ++e) c[i][j][e] = 0.0f;

    // ---- prologue: stage tile 0 into buffer 0 ----
    if constexpr (!TRANSA)
        stage_tile_direct(A + (size_t)m0 * lda, lda, Mreal, As[0], tid);
    else
        stage_tile_transpose(A + m0, lda, m0, Mreal, As[0], tid);
    stage_tile_direct(BT + (size_t)n0 * ldbt, ldbt, N, Bs[0], tid);

    int pb = 0;
    for (int kt = 0; kt < K; kt += BK, pb ^= 1) {
#if HAVE_TDM
        __builtin_amdgcn_s_wait_tensorcnt(0);   // in-flight tile complete
#endif
        __syncthreads();                        // publish tile kt / retire kt-BK reads

        const int ktn = kt + BK;
        if (ktn < K) {
            // stage tile kt+BK into the other buffer; overlaps compute below
            if constexpr (!TRANSA)
                stage_tile_direct(A + (size_t)m0 * lda + ktn, lda, Mreal, As[pb ^ 1], tid);
            else
                stage_tile_transpose(A + (size_t)ktn * lda + m0, lda, m0, Mreal,
                                     As[pb ^ 1], tid);
            stage_tile_direct(BT + (size_t)n0 * ldbt + ktn, ldbt, N, Bs[pb ^ 1], tid);
            if (ktn + BK < K) {   // L2 prefetch two tiles ahead
                __builtin_prefetch(TRANSA ? (const void*)(A + (size_t)(ktn + BK) * lda + m0)
                                          : (const void*)(A + (size_t)m0 * lda + ktn + BK), 0, 1);
                __builtin_prefetch((const void*)(BT + (size_t)n0 * ldbt + ktn + BK), 0, 1);
            }
        }

        __bf16 (*Ac)[LDSW] = As[pb];
        __bf16 (*Bc)[LDSW] = Bs[pb];

        // two K=32 sub-steps per staged tile -> 16 WMMAs per barrier
#pragma unroll
        for (int kk = 0; kk < BK; kk += 32) {
            v16bf af[2], bfv[4];
#pragma unroll
            for (int i = 0; i < 2; ++i) {
                int row = wr * 32 + i * 16 + l16;
                v8bf lo = *(const v8bf*)&Ac[row][kk + sel * 8];        // K 0..7  / 8..15
                v8bf hi = *(const v8bf*)&Ac[row][kk + 16 + sel * 8];   // K 16..23 / 24..31
                af[i] = __builtin_shufflevector(lo, hi, 0, 1, 2, 3, 4, 5, 6, 7,
                                                8, 9, 10, 11, 12, 13, 14, 15);
            }
#pragma unroll
            for (int j = 0; j < 4; ++j) {
                int row = wc * 64 + j * 16 + l16;
                v8bf lo = *(const v8bf*)&Bc[row][kk + sel * 8];
                v8bf hi = *(const v8bf*)&Bc[row][kk + 16 + sel * 8];
                bfv[j] = __builtin_shufflevector(lo, hi, 0, 1, 2, 3, 4, 5, 6, 7,
                                                 8, 9, 10, 11, 12, 13, 14, 15);
            }
#pragma unroll
            for (int i = 0; i < 2; ++i)
#pragma unroll
                for (int j = 0; j < 4; ++j)
                    c[i][j] = __builtin_amdgcn_wmma_f32_16x16x32_bf16(
                        false, af[i], false, bfv[j], (short)0, c[i][j], false, false);
        }
    }

    // ---- epilogue ----
#pragma unroll
    for (int i = 0; i < 2; ++i) {
#pragma unroll
        for (int j = 0; j < 4; ++j) {
            int gmb = m0 + wr * 32 + i * 16 + sel * 8;
            int gn  = n0 + wc * 64 + j * 16 + l16;
            if constexpr (MODE == 0) {
                float eta = vec[gn];
                v8bf pt;
#pragma unroll
                for (int r = 0; r < 8; ++r) {
                    int m = gmb + r;
                    size_t idx = (size_t)m * N + gn;
                    float st = c[i][j][r] * scl;                       // decay folded in
                    float me = (S[idx] - (float)spike[idx] * 0.5f) * 0.8f + st;
                    S[idx] = me;
                    spike[idx] = (__bf16)((me > 0.5f) ? 1.0f : 0.0f);
                    pt[r] = (__bf16)tanhf(me * 2.0f - eta);
                }
                // postT is N-major: row gn, cols gmb..gmb+7 contiguous (16B store)
                *(v8bf*)&postT[(size_t)gn * Mreal + gmb] = pt;
            } else {
#pragma unroll
                for (int r = 0; r < 8; ++r) {
                    int m = gmb + r;
                    if (m < Mreal) {
                        size_t idx = (size_t)m * N + gn;
                        S[idx] = 0.8f * S[idx] + vec[m] * scl * c[i][j][r];
                    }
                }
            }
        }
    }
}

// ------------------------------- final output ------------------------------
// out[b,o] = sum_k mem2[b,k] * fc3[o,k] * mask2[k,o]     (N=10, K=1024)
__global__ __launch_bounds__(256)
void final_out_kernel(const float* __restrict__ mem2, const float* __restrict__ fc3,
                      const float* __restrict__ mask2, float* __restrict__ out) {
    __shared__ float red[256];
    const int b = blockIdx.x;
    const int tid = threadIdx.x;
    float acc[10];
#pragma unroll
    for (int o = 0; o < 10; ++o) acc[o] = 0.0f;
    const float* row = mem2 + (size_t)b * 1024;
    for (int k = tid; k < 1024; k += 256) {
        float h = row[k];
#pragma unroll
        for (int o = 0; o < 10; ++o)
            acc[o] += h * fc3[o * 1024 + k] * mask2[k * 10 + o];
    }
    for (int o = 0; o < 10; ++o) {
        red[tid] = acc[o];
        __syncthreads();
        for (int s = 128; s > 0; s >>= 1) {
            if (tid < s) red[tid] += red[tid + s];
            __syncthreads();
        }
        if (tid == 0) out[(size_t)b * 10 + o] = red[0];
        __syncthreads();
    }
}

// ------------------------------- host launch -------------------------------

extern "C" void kernel_launch(void* const* d_in, const int* in_sizes, int n_in,
                              void* d_out, int out_size, void* d_ws, size_t ws_size,
                              hipStream_t stream) {
    constexpr int Bn  = 8192;
    constexpr int IN  = 784;
    constexpr int INP = 832;   // K padded to multiple of BK(=64)
    constexpr int H1  = 1024;
    constexpr int H2  = 1024;

    const float* x     = (const float*)d_in[0];
    const float* mask0 = (const float*)d_in[1];
    const float* mask1 = (const float*)d_in[2];
    const float* mask2 = (const float*)d_in[3];
    const float* fc1_w = (const float*)d_in[4];
    const float* fc2_w = (const float*)d_in[5];
    const float* fc3_w = (const float*)d_in[6];
    const float* alpha1 = (const float*)d_in[7];
    const float* alpha2 = (const float*)d_in[8];
    const float* beta1  = (const float*)d_in[9];
    const float* beta2  = (const float*)d_in[10];
    const float* eta1   = (const float*)d_in[11];
    const float* eta2   = (const float*)d_in[12];
    float* out = (float*)d_out;

    char* w = (char*)d_ws;
    auto carve = [&](size_t bytes) -> char* {
        char* p = w;
        w += (bytes + 255) & ~(size_t)255;
        return p;
    };
    __bf16* xb     = (__bf16*)carve((size_t)Bn * INP * 2);
    __bf16* s1b    = (__bf16*)carve((size_t)Bn * H1 * 2);
    __bf16* s2b    = (__bf16*)carve((size_t)Bn * H2 * 2);
    __bf16* post1T = (__bf16*)carve((size_t)H1 * Bn * 2);   // N-major (H1 x B)
    __bf16* post2T = (__bf16*)carve((size_t)H2 * Bn * 2);   // N-major (H2 x B)
    __bf16* fw1T   = (__bf16*)carve((size_t)H1 * INP * 2);  // N-major (H1 x INP)
    __bf16* fw2T   = (__bf16*)carve((size_t)H2 * H1 * 2);   // N-major (H2 x H1)
    float*  mem1   = (float*)carve((size_t)Bn * H1 * 4);
    float*  mem2   = (float*)carve((size_t)Bn * H2 * 4);
    float*  hebb1  = (float*)carve((size_t)IN * H1 * 4);
    float*  hebb2  = (float*)carve((size_t)H1 * H2 * 4);

    auto zero = [&](void* p, size_t bytes) {
        long n16 = (long)(bytes / 16);
        fill0_kernel<<<dim3((unsigned)((n16 + 255) / 256)), dim3(256), 0, stream>>>(
            (uint4*)p, n16);
    };
    zero(mem1, (size_t)Bn * H1 * 4);
    zero(mem2, (size_t)Bn * H2 * 4);
    zero(hebb1, (size_t)IN * H1 * 4);
    zero(hebb2, (size_t)H1 * H2 * 4);
    zero(s1b, (size_t)Bn * H1 * 2);
    zero(s2b, (size_t)Bn * H2 * 2);

    {
        int total = Bn * INP;
        cvt_x_kernel<<<dim3((total + 255) / 256), dim3(256), 0, stream>>>(x, xb, IN, INP, total);
    }

    const float dec[3] = {1.0f, expf(-1.0f / 50.0f), expf(-2.0f / 50.0f)};
    const float invB = 1.0f / (float)Bn;

    for (int step = 0; step < 3; ++step) {
        const float decay = dec[step];

        // ----- layer 1 -----
        prep_fastwT_kernel<<<dim3((H1 * INP + 255) / 256), dim3(256), 0, stream>>>(
            fc1_w, mask0, hebb1, alpha1, fw1T, IN, INP, H1);
        gemm_wmma_kernel<false, 0><<<dim3(H1 / BN, Bn / BM), dim3(256), 0, stream>>>(
            xb, INP, fw1T, INP, INP, mem1, s1b, post1T, eta1, decay, Bn, H1);
        gemm_wmma_kernel<true, 1><<<dim3(H1 / BN, (IN + BM - 1) / BM), dim3(256), 0, stream>>>(
            xb, INP, post1T, Bn, Bn, hebb1, nullptr, nullptr, beta1, decay * invB, IN, H1);

        // ----- layer 2 -----
        prep_fastwT_kernel<<<dim3((H2 * H1 + 255) / 256), dim3(256), 0, stream>>>(
            fc2_w, mask1, hebb2, alpha2, fw2T, H1, H1, H2);
        gemm_wmma_kernel<false, 0><<<dim3(H2 / BN, Bn / BM), dim3(256), 0, stream>>>(
            s1b, H1, fw2T, H1, H1, mem2, s2b, post2T, eta2, decay, Bn, H2);
        gemm_wmma_kernel<true, 1><<<dim3(H2 / BN, H1 / BM), dim3(256), 0, stream>>>(
            s1b, H1, post2T, Bn, Bn, hebb2, nullptr, nullptr, beta2, decay * invB, H1, H2);
    }

    final_out_kernel<<<dim3(Bn), dim3(256), 0, stream>>>(mem2, fc3_w, mask2, out);
}
